// MoEMLP_86234353369530
// MI455X (gfx1250) — compile-verified, matrix operand order checked by reading, not run on hardware
//
#include <hip/hip_runtime.h>
#include <cmath>

// ---------- types ----------
typedef __attribute__((ext_vector_type(16))) __bf16       v16bf;
typedef __attribute__((ext_vector_type(8)))  float        v8f;
typedef __attribute__((ext_vector_type(4)))  float        v4f;
typedef __attribute__((ext_vector_type(4)))  unsigned int v4u;
typedef __attribute__((ext_vector_type(2)))  unsigned int v2u;

union Frag { v16bf v; v4u q[2]; };   // 32 bytes: one WMMA bf16 A/B operand

// CDNA5 async global->LDS path (ASYNCcnt) if this toolchain exposes it.
#if defined(__has_builtin)
#if __has_builtin(__builtin_amdgcn_global_load_async_to_lds_b128) && \
    __has_builtin(__builtin_amdgcn_s_wait_asynccnt)
#define HAVE_ASYNC_LDS 1
#endif
#endif

#ifdef HAVE_ASYNC_LDS
// Builtin signature (from hipcc diagnostic): arg0 = v4int* in address_space(1),
// arg1 = v4int* in address_space(3), then imm offset, imm cpol.
typedef int v4int __attribute__((vector_size(16)));
typedef __attribute__((address_space(1))) v4int* gv4p;
typedef __attribute__((address_space(3))) v4int* lv4p;
#endif

// ---------- problem constants ----------
constexpr int NTOK   = 4096;          // B*T
constexpr int HD     = 1024;          // H
constexpr int FD     = 4096;          // F
constexpr int NE     = 8;             // experts
constexpr int TOPK   = 2;
constexpr int NSLOT  = NTOK * TOPK;       // 8192
constexpr int PADMAX = NSLOT + NE * 16;   // 8320 (per-expert 16-padding)
constexpr int MAXTIL = PADMAX / 16;       // 520 tiles of 16 rows

// ---------- small helpers ----------
__global__ void zero_meta(int* counts, float* sumP, int* fillptr) {
  int i = threadIdx.x;
  if (i < NE) { counts[i] = 0; sumP[i] = 0.f; fillptr[i] = 0; }
}

// fp32 -> bf16, 4 elems/thread
__global__ __launch_bounds__(256) void cvt_hidden(const float* __restrict__ in,
                                                  __bf16* __restrict__ out) {
  int i = (blockIdx.x * 256 + threadIdx.x) * 4;
  v4f x = *(const v4f*)(in + i);
  union { __bf16 b[4]; v2u u; } r;
  r.b[0] = (__bf16)x.x; r.b[1] = (__bf16)x.y;
  r.b[2] = (__bf16)x.z; r.b[3] = (__bf16)x.w;
  *(v2u*)(out + i) = r.u;
}

// in [e][R][C] fp32  ->  out [e][C][R] bf16  (LDS-tiled 32x32 transpose)
__global__ __launch_bounds__(256) void transpose_cvt(const float* __restrict__ in,
                                                     __bf16* __restrict__ out,
                                                     int R, int C) {
  __shared__ float tile[32][33];
  int e  = blockIdx.z;
  int c0 = blockIdx.x * 32, r0 = blockIdx.y * 32;
  int tx = threadIdx.x & 31, ty = threadIdx.x >> 5;   // 32 x 8
  const float* src = in  + (size_t)e * R * C;
  __bf16*      dst = out + (size_t)e * R * C;
#pragma unroll
  for (int j = 0; j < 4; ++j)
    tile[ty + j * 8][tx] = src[(size_t)(r0 + ty + j * 8) * C + (c0 + tx)];
  __syncthreads();
#pragma unroll
  for (int j = 0; j < 4; ++j)
    dst[(size_t)(c0 + ty + j * 8) * R + (r0 + tx)] = (__bf16)tile[tx][ty + j * 8];
}

// ---------- router: 1 wave per token ----------
__global__ __launch_bounds__(256) void router_kernel(const float* __restrict__ hs,
                                                     const float* __restrict__ rw,
                                                     int* __restrict__ topk_idx,
                                                     float* __restrict__ topk_w,
                                                     int* __restrict__ counts,
                                                     float* __restrict__ sumP) {
  int t    = blockIdx.x * 8 + (threadIdx.x >> 5);
  int lane = threadIdx.x & 31;
  if (t >= NTOK) return;
  float p[NE];
#pragma unroll
  for (int e = 0; e < NE; ++e) p[e] = 0.f;
  const float* x = hs + (size_t)t * HD;
  for (int h = lane; h < HD; h += 32) {
    float xv = x[h];
#pragma unroll
    for (int e = 0; e < NE; ++e) p[e] += xv * rw[e * HD + h];
  }
#pragma unroll
  for (int e = 0; e < NE; ++e)
    for (int off = 16; off; off >>= 1) p[e] += __shfl_down(p[e], off, 32);
  if (lane == 0) {
    int i0 = 0; float v0 = p[0];
    for (int e = 1; e < NE; ++e) if (p[e] > v0) { v0 = p[e]; i0 = e; }
    int i1 = (i0 == 0) ? 1 : 0; float v1 = p[i1];
    for (int e = 0; e < NE; ++e) if (e != i0 && p[e] > v1) { v1 = p[e]; i1 = e; }
    float m2 = fmaxf(v0, v1);
    float e0 = expf(v0 - m2), e1 = expf(v1 - m2), s = e0 + e1;
    topk_idx[t * 2] = i0;  topk_idx[t * 2 + 1] = i1;
    topk_w[t * 2] = e0 / s; topk_w[t * 2 + 1] = e1 / s;
    atomicAdd(&counts[i0], 1); atomicAdd(&counts[i1], 1);
    float mm = p[0];
    for (int e = 1; e < NE; ++e) mm = fmaxf(mm, p[e]);
    float se = 0.f, ex[NE];
    for (int e = 0; e < NE; ++e) { ex[e] = expf(p[e] - mm); se += ex[e]; }
    for (int e = 0; e < NE; ++e) atomicAdd(&sumP[e], ex[e] / se);
  }
}

// ---------- scan: padded offsets + aux loss ----------
__global__ void scan_kernel(const int* __restrict__ counts, const float* __restrict__ sumP,
                            int* __restrict__ poff, int* __restrict__ fillptr,
                            int* __restrict__ totalp, float* __restrict__ out_scalars) {
  if (threadIdx.x == 0 && blockIdx.x == 0) {
    int acc = 0; float aux = 0.f;
    for (int e = 0; e < NE; ++e) {
      poff[e] = acc;
      int c = counts[e];
      acc += (c + 15) & ~15;
      fillptr[e] = 0;
      float f = (float)c / (float)(NSLOT);
      float P = sumP[e] / (float)NTOK;
      aux += f * P;
    }
    poff[NE] = acc;
    *totalp  = acc;
    out_scalars[0] = (float)NE * aux;  // aux_loss
    out_scalars[1] = 0.f;              // statement_loss
  }
}

__global__ void init_perm(int* perm) {
  int i = blockIdx.x * 256 + threadIdx.x;
  if (i < PADMAX) perm[i] = -1;
}

__global__ void scatter_kernel(const int* __restrict__ topk_idx, const int* __restrict__ poff,
                               int* __restrict__ fillptr, int* __restrict__ perm) {
  int i = blockIdx.x * 256 + threadIdx.x;   // slot id = token*2 + k
  int e = topk_idx[i];
  int pos = atomicAdd(&fillptr[e], 1);
  perm[poff[e] + pos] = i;
}

// ---------- fused expert MLP over one 16-slot tile ----------
// 4 waves; wave w owns mid cols [w*16,w*16+16) in layer1 and out cols [w*256,(w+1)*256) in layer2.
__global__ __launch_bounds__(128) void moe_fused(const __bf16* __restrict__ hsb,
                                                 const __bf16* __restrict__ w1t, // [e][f][h]
                                                 const __bf16* __restrict__ w2t, // [e][h][f]
                                                 const float* __restrict__ b1,   // [e][f]
                                                 const int* __restrict__ perm,
                                                 const int* __restrict__ poff,
                                                 const int* __restrict__ totalp,
                                                 float* __restrict__ outslot) {
  __shared__ __align__(16) __bf16 Xs[16][HD];   // 32 KB token tile (bf16)
  __shared__ __align__(16) __bf16 Ms[16][64];   // 2 KB mid chunk (bf16)
  __shared__ int rowslot[16];

  int tile = blockIdx.x;
  int base = tile * 16;
  if (base >= *totalp) return;

  int e = 0;
  for (int i = 1; i < NE; ++i) if (base >= poff[i]) e = i;

  int tid = threadIdx.x;
  if (tid < 16) rowslot[tid] = perm[base + tid];
  __syncthreads();

  { // stage X tile: 8 threads per row, 128-bit chunks
    int m = tid >> 3, c0 = tid & 7;
    int slot = rowslot[m];
#ifdef HAVE_ASYNC_LDS
    if (slot >= 0) {
      gv4p src = (gv4p)(hsb + (size_t)(slot >> 1) * HD);
      lv4p dst = (lv4p)&Xs[m][0];
#pragma unroll
      for (int c = 0; c < 16; ++c) {
        int idx = c0 + c * 8;   // 16-byte chunk index within the row
        __builtin_amdgcn_global_load_async_to_lds_b128(src + idx, dst + idx, 0, 0);
      }
    } else {
      v4u z = (v4u)0u;
      v4u* dst = (v4u*)&Xs[m][0];
      for (int c = c0; c < HD / 8; c += 8) dst[c] = z;
    }
    __builtin_amdgcn_s_wait_asynccnt(0);
#else
    v4u* dst = (v4u*)&Xs[m][0];
    if (slot >= 0) {
      const v4u* src = (const v4u*)(hsb + (size_t)(slot >> 1) * HD);
      for (int c = c0; c < HD / 8; c += 8) dst[c] = src[c];
    } else {
      v4u z = (v4u)0u;
      for (int c = c0; c < HD / 8; c += 8) dst[c] = z;
    }
#endif
  }
  __syncthreads();

  int lane  = tid & 31, w = tid >> 5;
  int nIdx  = lane & 15;
  int kBase = (lane >> 4) << 3;        // 0 or 8 (per ISA 16-bit A/B lane split)
  int mRow  = (lane >> 4) << 3;        // C/D: m = r + 8*(lane>=16)

  const __bf16* w1e = w1t + (size_t)e * FD * HD;
  const __bf16* w2e = w2t + (size_t)e * HD * FD;
  const float*  b1e = b1 + e * FD;

  v8f acc[16];
#pragma unroll
  for (int i = 0; i < 16; ++i)
#pragma unroll
    for (int r = 0; r < 8; ++r) acc[i][r] = 0.f;

  for (int fc = 0; fc < FD / 64; ++fc) {
    // ---- layer 1: mid[:, w*16+nIdx] over full H ----
    int fcol = fc * 64 + w * 16 + nIdx;
    float bb = b1e[fcol];
    v8f a1;
#pragma unroll
    for (int r = 0; r < 8; ++r) a1[r] = bb;

    const v4u* aptr = (const v4u*)(&Xs[nIdx][kBase]);
    const v4u* bptr = (const v4u*)(w1e + (size_t)fcol * HD + kBase);
    if (fc + 1 < FD / 64)  // warm L2 for next chunk's W1 column block
      __builtin_prefetch(w1e + (size_t)(fcol + 64) * HD, 0, 1);
#pragma unroll 4
    for (int h0 = 0; h0 < HD; h0 += 32) {
      Frag A, B;
      A.q[0] = aptr[0]; A.q[1] = aptr[2];   // k {0..7}+kBase, {16..23}+kBase
      B.q[0] = bptr[0]; B.q[1] = bptr[2];
      aptr += 4; bptr += 4;
      a1 = __builtin_amdgcn_wmma_f32_16x16x32_bf16(false, A.v, false, B.v,
                                                   (short)0, a1, false, false);
    }
    // exact GELU, stash mid chunk as bf16
#pragma unroll
    for (int r = 0; r < 8; ++r) {
      float x = a1[r];
      float g = 0.5f * x * (1.0f + erff(x * 0.70710678118654752f));
      Ms[mRow + r][w * 16 + nIdx] = (__bf16)g;
    }
    __syncthreads();

    // ---- layer 2: out[:, w*256 .. w*256+255] += mid_chunk @ W2 ----
    Frag A0, A1;
    {
      const v4u* mp = (const v4u*)(&Ms[nIdx][kBase]);
      A0.q[0] = mp[0]; A0.q[1] = mp[2];    // k 0..31 of chunk
      A1.q[0] = mp[4]; A1.q[1] = mp[6];    // k 32..63 of chunk
    }
#pragma unroll
    for (int nt = 0; nt < 16; ++nt) {
      int ncol = w * 256 + nt * 16 + nIdx;
      const v4u* bp = (const v4u*)(w2e + (size_t)ncol * FD + fc * 64 + kBase);
      Frag B0, B1;
      B0.q[0] = bp[0]; B0.q[1] = bp[2];
      B1.q[0] = bp[4]; B1.q[1] = bp[6];
      acc[nt] = __builtin_amdgcn_wmma_f32_16x16x32_bf16(false, A0.v, false, B0.v,
                                                        (short)0, acc[nt], false, false);
      acc[nt] = __builtin_amdgcn_wmma_f32_16x16x32_bf16(false, A1.v, false, B1.v,
                                                        (short)0, acc[nt], false, false);
    }
    __syncthreads();
  }

  // per-slot store (no atomics -> deterministic combine)
#pragma unroll
  for (int r = 0; r < 8; ++r) {
    int slot = rowslot[mRow + r];
    if (slot >= 0) {
#pragma unroll
      for (int nt = 0; nt < 16; ++nt)
        outslot[(size_t)slot * HD + w * 256 + nt * 16 + nIdx] = acc[nt][r];
    }
  }
}

// ---------- combine: weighted top-k mix + b2 ----------
__global__ __launch_bounds__(256) void combine_kernel(const float* __restrict__ outslot,
                                                      const int* __restrict__ topk_idx,
                                                      const float* __restrict__ topk_w,
                                                      const float* __restrict__ b2,
                                                      float* __restrict__ out) {
  int i = blockIdx.x * 256 + threadIdx.x;   // NTOK*HD threads
  int t = i >> 10, h = i & (HD - 1);
  int e0 = topk_idx[t * 2], e1 = topk_idx[t * 2 + 1];
  float w0 = topk_w[t * 2], w1v = topk_w[t * 2 + 1];
  float r = w0  * (outslot[(size_t)(t * 2) * HD + h]     + b2[e0 * HD + h])
          + w1v * (outslot[(size_t)(t * 2 + 1) * HD + h] + b2[e1 * HD + h]);
  out[i] = r;
}

// ---------- launch ----------
extern "C" void kernel_launch(void* const* d_in, const int* in_sizes, int n_in,
                              void* d_out, int out_size, void* d_ws, size_t ws_size,
                              hipStream_t stream) {
  (void)in_sizes; (void)n_in; (void)out_size; (void)ws_size;
  const float* hs = (const float*)d_in[0];
  const float* rw = (const float*)d_in[1];
  const float* w1 = (const float*)d_in[2];
  const float* b1 = (const float*)d_in[3];
  const float* w2 = (const float*)d_in[4];
  const float* b2 = (const float*)d_in[5];
  float* out = (float*)d_out;

  // workspace layout (~176.5 MB): hsb 8M | w1t 64M | w2t 64M | outslot 32M | meta
  char* ws = (char*)d_ws;
  __bf16* hsb   = (__bf16*)ws;
  __bf16* w1t   = (__bf16*)(ws + (8ull) * 1024 * 1024);
  __bf16* w2t   = (__bf16*)(ws + (8ull + 64) * 1024 * 1024);
  float*  oslot = (float*)(ws + (8ull + 64 + 64) * 1024 * 1024);
  int*    meta  = (int*)  (ws + (8ull + 64 + 64 + 32) * 1024 * 1024);

  int*   topk_idx = meta;
  float* topk_w   = (float*)(meta + 8192);
  int*   counts   = meta + 16384;
  float* sumP     = (float*)(meta + 16392);
  int*   poff     = meta + 16400;
  int*   fillptr  = meta + 16409;
  int*   totalp   = meta + 16417;
  int*   perm     = meta + 16420;

  zero_meta<<<1, 32, 0, stream>>>(counts, sumP, fillptr);
  cvt_hidden<<<(NTOK * HD) / (256 * 4), 256, 0, stream>>>(hs, hsb);
  transpose_cvt<<<dim3(FD / 32, HD / 32, NE), 256, 0, stream>>>(w1, w1t, HD, FD);
  transpose_cvt<<<dim3(HD / 32, FD / 32, NE), 256, 0, stream>>>(w2, w2t, FD, HD);
  router_kernel<<<NTOK / 8, 256, 0, stream>>>(hs, rw, topk_idx, topk_w, counts, sumP);
  scan_kernel<<<1, 1, 0, stream>>>(counts, sumP, poff, fillptr, totalp,
                                   out + (size_t)NTOK * HD);
  init_perm<<<(PADMAX + 255) / 256, 256, 0, stream>>>(perm);
  scatter_kernel<<<NSLOT / 256, 256, 0, stream>>>(topk_idx, poff, fillptr, perm);
  moe_fused<<<MAXTIL, 128, 0, stream>>>(hsb, w1t, w2t, b1, perm, poff, totalp, oslot);
  combine_kernel<<<(NTOK * HD) / 256, 256, 0, stream>>>(oslot, topk_idx, topk_w, b2, out);
}
